// TransformerBlock_78778290144122
// MI455X (gfx1250) — matrix operand order repaired and checked by reading, not compiled, not run on hardware
//
#include <hip/hip_runtime.h>
#include <hip/hip_bf16.h>
#include <stdint.h>

#define DIMN 1024
#define HIDN 4096
#define NHN  16
#define HDN  64
#define BN   4
#define SN   2048
#define ROWS (BN * SN)

#define TM 128
#define TN 128
#define TK 32
#define LSA 40   // LDS stride in bf16 units (80B = 64B row + 16B TDM pad)
#define VSA 72   // V LDS stride in bf16 units (144B = 128B row + 16B TDM pad)

typedef __attribute__((ext_vector_type(16))) __bf16 bf16x16;
typedef __attribute__((ext_vector_type(8)))  float  f32x8;
typedef __attribute__((ext_vector_type(4)))  unsigned int u32x4;
typedef __attribute__((ext_vector_type(8)))  int    i32x8;

__device__ __forceinline__ bf16x16 pack16(uint4 lo, uint4 hi) {
    union { uint4 q[2]; bf16x16 v; } u;
    u.q[0] = lo; u.q[1] = hi;
    return u.v;
}

// ---------------------------------------------------------------------------
// DPP16 reductions within each 16-lane half of a wave32 (no LDS, no waits).
// xor1/xor2 via quad_perm, then mirror patterns act as xor4/xor8 once the
// smaller groups are uniform. DPP never crosses the 16-lane row boundary,
// so lanes 0-15 and 16-31 reduce independently (exactly the C-matrix halves).
// ---------------------------------------------------------------------------
__device__ __forceinline__ float red_max16(float v) {
    v = fmaxf(v, __int_as_float(__builtin_amdgcn_update_dpp(
                     0, __float_as_int(v), 0x0B1, 0xF, 0xF, true)));  // quad_perm xor1
    v = fmaxf(v, __int_as_float(__builtin_amdgcn_update_dpp(
                     0, __float_as_int(v), 0x04E, 0xF, 0xF, true)));  // quad_perm xor2
    v = fmaxf(v, __int_as_float(__builtin_amdgcn_update_dpp(
                     0, __float_as_int(v), 0x141, 0xF, 0xF, true)));  // row_half_mirror
    v = fmaxf(v, __int_as_float(__builtin_amdgcn_update_dpp(
                     0, __float_as_int(v), 0x140, 0xF, 0xF, true)));  // row_mirror
    return v;
}
__device__ __forceinline__ float red_add16(float v) {
    v += __int_as_float(__builtin_amdgcn_update_dpp(0, __float_as_int(v), 0x0B1, 0xF, 0xF, true));
    v += __int_as_float(__builtin_amdgcn_update_dpp(0, __float_as_int(v), 0x04E, 0xF, 0xF, true));
    v += __int_as_float(__builtin_amdgcn_update_dpp(0, __float_as_int(v), 0x141, 0xF, 0xF, true));
    v += __int_as_float(__builtin_amdgcn_update_dpp(0, __float_as_int(v), 0x140, 0xF, 0xF, true));
    return v;
}

// ---------------------------------------------------------------------------
// Tensor Data Mover: async 2D tile load global -> LDS with row padding.
// D# packed per CDNA5 ISA (group0 128b + group1 256b, 2D
// form with VADDR2/3 = NULL). Tracked by TENSORcnt.
// ---------------------------------------------------------------------------
__device__ __forceinline__ void tdm_load_2d(const void* lds_dst, const void* gsrc,
                                            unsigned tile_d0, unsigned tile_d1,
                                            unsigned long long tens_d0,
                                            unsigned long long tens_d1,
                                            unsigned long long stride0,
                                            unsigned pad_int_code,
                                            unsigned pad_amt_code) {
    const unsigned long long ga = (unsigned long long)(uintptr_t)gsrc;
    const unsigned lds_off = (unsigned)(uintptr_t)lds_dst;  // low 32b = LDS offset
    u32x4 g0;
    g0.x = 1u;                                               // count=1 (valid user D#)
    g0.y = lds_off;                                          // lds_addr [63:32]
    g0.z = (unsigned)ga;                                     // global_addr [95:64]
    g0.w = (unsigned)((ga >> 32) & 0x01ffffffu) | (2u << 30);// addr[56:32], type=2
    i32x8 g1;
    g1[0] = (int)((1u << 16)                                 // data_size = 2 bytes
                | (1u << 20)                                 // pad_enable
                | ((pad_int_code & 7u) << 22)
                | ((pad_amt_code & 0x7fu) << 25));
    g1[1] = (int)((unsigned)(tens_d0 & 0xffffu) << 16);      // tensor_dim0[15:0]
    g1[2] = (int)(((unsigned)(tens_d0 >> 16) & 0xffffu)      // tensor_dim0[31:16]
                | ((unsigned)(tens_d1 & 0xffffu) << 16));    // tensor_dim1[15:0]
    g1[3] = (int)(((unsigned)(tens_d1 >> 16) & 0xffffu)      // tensor_dim1[31:16]
                | ((tile_d0 & 0xffffu) << 16));              // tile_dim0
    g1[4] = (int)(tile_d1 & 0xffffu);                        // tile_dim1 (tile_dim2=0)
    g1[5] = (int)(unsigned)(stride0 & 0xffffffffu);          // tensor_dim0_stride lo
    g1[6] = (int)((unsigned)(stride0 >> 32) & 0xffffu);      // stride hi (dim1_stride=0)
    g1[7] = 0;
    asm volatile("tensor_load_to_lds %0, %1" :: "s"(g0), "s"(g1) : "memory");
}

// ---------------------------------------------------------------------------
// LayerNorm (fp32 in) -> bf16 out. One block per row of 1024.
// ---------------------------------------------------------------------------
__global__ __launch_bounds__(256) void ln_bf16(const float* __restrict__ x,
                                               const float* __restrict__ g,
                                               const float* __restrict__ be,
                                               __bf16* __restrict__ y) {
    __shared__ float s1[256];
    __shared__ float s2[256];
    const int row = blockIdx.x;
    const int t   = threadIdx.x;
    const float4 d = ((const float4*)(x + (size_t)row * DIMN))[t];
    s1[t] = d.x + d.y + d.z + d.w;
    s2[t] = d.x * d.x + d.y * d.y + d.z * d.z + d.w * d.w;
    __syncthreads();
    for (int o = 128; o > 0; o >>= 1) {
        if (t < o) { s1[t] += s1[t + o]; s2[t] += s2[t + o]; }
        __syncthreads();
    }
    const float mu  = s1[0] * (1.0f / DIMN);
    const float var = s2[0] * (1.0f / DIMN) - mu * mu;
    const float rs  = rsqrtf(var + 1e-5f);
    const float4 gg = ((const float4*)g)[t];
    const float4 bb = ((const float4*)be)[t];
    __bf16* yr = y + (size_t)row * DIMN + t * 4;
    yr[0] = (__bf16)((d.x - mu) * rs * gg.x + bb.x);
    yr[1] = (__bf16)((d.y - mu) * rs * gg.y + bb.y);
    yr[2] = (__bf16)((d.z - mu) * rs * gg.z + bb.z);
    yr[3] = (__bf16)((d.w - mu) * rs * gg.w + bb.w);
}

// ---------------------------------------------------------------------------
// B-tile staging: 32x128 fp32 -> bf16, transposed into Bs[n][k].
// ---------------------------------------------------------------------------
__device__ __forceinline__ void stage_b(__bf16* bsd, const float* __restrict__ Bw,
                                        int N, int kkk, int n0, int bkr, int bseg) {
    const float4* bg = (const float4*)(Bw + (size_t)(kkk + bkr) * N + n0 + bseg * 16);
    float4 f0 = bg[0], f1 = bg[1], f2 = bg[2], f3 = bg[3];
    float tf[16] = { f0.x, f0.y, f0.z, f0.w, f1.x, f1.y, f1.z, f1.w,
                     f2.x, f2.y, f2.z, f2.w, f3.x, f3.y, f3.z, f3.w };
    #pragma unroll
    for (int j = 0; j < 16; ++j)
        bsd[(bseg * 16 + j) * LSA + bkr] = (__bf16)tf[j];
}

// ---------------------------------------------------------------------------
// bf16 GEMM, f32 accumulate via v_wmma_f32_16x16x32_bf16.
// Double-buffered LDS: while tile k is consumed by the WMMAs, the Tensor Data
// Mover streams the A tile for k+1 (wave 0 issues/waits TENSORcnt) and the
// ALUs stage/convert the fp32 B tile for k+1.
// EPI: 0 = bf16 out, 1 = ReLU -> bf16 out, 2 = fp32 out with residual add.
// ---------------------------------------------------------------------------
template <int EPI>
__global__ __launch_bounds__(256) void gemm_bf16(const __bf16* __restrict__ A,
                                                 const float* __restrict__ Bw,
                                                 const float* __restrict__ resid,
                                                 void* __restrict__ Cout,
                                                 int M, int N, int K) {
    __shared__ __bf16 As[2][TM * LSA];
    __shared__ __bf16 Bs[2][TN * LSA];   // transposed: Bs[n][k]

    const int tid  = threadIdx.x;
    const int wave = tid >> 5;
    const int lane = tid & 31;
    const int lh   = lane >> 4;
    const int ln   = lane & 15;
    const int wm   = wave & 3;        // 4 row-waves of 32
    const int wn   = wave >> 2;       // 2 col-waves of 64
    const int m0   = blockIdx.y * TM;
    const int n0   = blockIdx.x * TN;

    f32x8 acc[2][4] = {};
    const int bkr = tid & 31, bseg = tid >> 5;

    // ---- prologue: fill buffer 0
    if (wave == 0)
        tdm_load_2d(As[0], A + (size_t)m0 * K, TK, TM,
                    (unsigned long long)K, (unsigned long long)M,
                    (unsigned long long)K, 3u, 3u);
    stage_b(Bs[0], Bw, N, 0, n0, bkr, bseg);
    if (wave == 0) __builtin_amdgcn_s_wait_tensorcnt(0);
    __syncthreads();

    int p = 0;
    for (int kk = 0; kk < K; kk += TK) {
        const int nxt = kk + TK;
        if (nxt < K) {   // overlap: fetch tile k+1 into the other buffer
            if (wave == 0)
                tdm_load_2d(As[p ^ 1], A + (size_t)m0 * K + nxt, TK, TM,
                            (unsigned long long)K, (unsigned long long)M,
                            (unsigned long long)K, 3u, 3u);
            stage_b(Bs[p ^ 1], Bw, N, nxt, n0, bkr, bseg);
            if (nxt + TK < K)
                __builtin_prefetch(Bw + (size_t)(nxt + TK + bkr) * N + n0 + bseg * 16, 0, 0);
        }

        // A fragments: lane ln = row, lh picks K-subgroup {0..7,16..23}/{8..15,24..31}
        const __bf16* asrc = As[p];
        const __bf16* bsrc = Bs[p];
        bf16x16 af[2];
        #pragma unroll
        for (int i = 0; i < 2; ++i) {
            const int r = wm * 32 + i * 16 + ln;
            uint4 lo = *(const uint4*)&asrc[r * LSA + lh * 8];
            uint4 hi = *(const uint4*)&asrc[r * LSA + 16 + lh * 8];
            af[i] = pack16(lo, hi);
        }
        // B fragments: lane ln = column, lh picks K-subgroup {0..7,16..23}/{8..15,24..31}
        bf16x16 bf[4];
        #pragma unroll
        for (int j = 0; j < 4; ++j) {
            const int c = wn * 64 + j * 16 + ln;
            uint4 lo = *(const uint4*)&bsrc[c * LSA + lh * 8];
            uint4 hi = *(const uint4*)&bsrc[c * LSA + 16 + lh * 8];
            bf[j] = pack16(lo, hi);
        }
        #pragma unroll
        for (int i = 0; i < 2; ++i)
            #pragma unroll
            for (int j = 0; j < 4; ++j)
                acc[i][j] = __builtin_amdgcn_wmma_f32_16x16x32_bf16(
                    false, af[i], false, bf[j], (short)0, acc[i][j], false, false);

        if (wave == 0 && nxt < K) __builtin_amdgcn_s_wait_tensorcnt(0);
        __syncthreads();
        p ^= 1;
    }

    // Epilogue. C layout: VGPR r -> row (r + 8*lh), col = ln within 16x16 tile.
    #pragma unroll
    for (int i = 0; i < 2; ++i) {
        const int rowb = m0 + wm * 32 + i * 16 + lh * 8;
        #pragma unroll
        for (int j = 0; j < 4; ++j) {
            const int colg = n0 + wn * 64 + j * 16 + ln;
            #pragma unroll
            for (int r = 0; r < 8; ++r) {
                const float v = acc[i][j][r];
                const size_t idx = (size_t)(rowb + r) * N + colg;
                if (EPI == 0)      ((__bf16*)Cout)[idx] = (__bf16)v;
                else if (EPI == 1) ((__bf16*)Cout)[idx] = (__bf16)(v > 0.f ? v : 0.f);
                else               ((float*)Cout)[idx] = resid[idx] + v;
            }
        }
    }
}

// ---------------------------------------------------------------------------
// Causal flash attention. One wave (32 thr) per (b, head, 16-query tile).
// Q,K,V,O: [B*S, DIM] bf16, head h at columns h*64..h*64+63.
// V tiles stream through a double-buffered LDS region via the TDM: block k+1
// is issued at the top of iteration k and waited (TENSORcnt<=1, in-order)
// only right before the P*V gather. Softmax row reductions use DPP16.
// ---------------------------------------------------------------------------
__global__ __launch_bounds__(32) void attn_flash(const __bf16* __restrict__ Q,
                                                 const __bf16* __restrict__ Kb,
                                                 const __bf16* __restrict__ Vb,
                                                 __bf16* __restrict__ O) {
    __shared__ __bf16 Pl[16 * LSA];      // P bounce buffer (16x32 tile)
    __shared__ __bf16 Vs[2][32 * VSA];   // V blocks, rows = key, stride 72 (TDM pad)

    const int lane = threadIdx.x;
    const int lh = lane >> 4, ln = lane & 15;
    const int q0 = blockIdx.x * 16;
    const int h  = blockIdx.y;
    const int b  = blockIdx.z;
    const size_t base = ((size_t)b * SN) * DIMN + h * HDN;

    // Q A-fragments for rows q0..q0+15, two K-steps over HD=64
    bf16x16 qf[2];
    #pragma unroll
    for (int ks = 0; ks < 2; ++ks) {
        const __bf16* qp = Q + base + (size_t)(q0 + ln) * DIMN + ks * 32 + lh * 8;
        qf[ks] = pack16(*(const uint4*)qp, *(const uint4*)(qp + 16));
    }

    float mrow[8], lrow[8];
    f32x8 o_[4] = {};
    #pragma unroll
    for (int r = 0; r < 8; ++r) { mrow[r] = -3.0e38f; lrow[r] = 0.f; }

    const int kend = q0 + 16;
    // prologue: TDM the first V block
    tdm_load_2d(Vs[0], Vb + base, HDN, 32,
                (unsigned long long)DIMN, (unsigned long long)SN,
                (unsigned long long)DIMN, 4u, 3u);

    int p = 0;
    for (int k0 = 0; k0 < kend; k0 += 32) {
        const bool more = (k0 + 32) < kend;
        if (more)   // overlap: prefetch next V block into the other buffer
            tdm_load_2d(Vs[p ^ 1], Vb + base + (size_t)(k0 + 32) * DIMN, HDN, 32,
                        (unsigned long long)DIMN, (unsigned long long)SN,
                        (unsigned long long)DIMN, 4u, 3u);

        // ---- scores: 16x32 = 2 WMMAs x 2 K-steps; K fragments direct from global
        f32x8 sc[2] = {};
        #pragma unroll
        for (int nt = 0; nt < 2; ++nt)
            #pragma unroll
            for (int ks = 0; ks < 2; ++ks) {
                const __bf16* kp =
                    Kb + base + (size_t)(k0 + nt * 16 + ln) * DIMN + ks * 32 + lh * 8;
                bf16x16 kf = pack16(*(const uint4*)kp, *(const uint4*)(kp + 16));
                sc[nt] = __builtin_amdgcn_wmma_f32_16x16x32_bf16(
                    false, qf[ks], false, kf, (short)0, sc[nt], false, false);
            }

        // ---- mask + online softmax; row reductions via DPP16 (per 16-lane half)
        float pv[2][8];
        #pragma unroll
        for (int r = 0; r < 8; ++r) {
            const int qi = q0 + r + 8 * lh;
            float best = -3.0e38f;
            #pragma unroll
            for (int nt = 0; nt < 2; ++nt) {
                const int kj = k0 + nt * 16 + ln;
                float sv = sc[nt][r] * 0.125f + ((kj > qi) ? -1.0e9f : 0.f);
                pv[nt][r] = sv;
                best = fmaxf(best, sv);
            }
            best = red_max16(best);
            const float mn    = fmaxf(mrow[r], best);
            const float alpha = __expf(mrow[r] - mn);
            mrow[r] = mn;
            float rs = 0.f;
            #pragma unroll
            for (int nt = 0; nt < 2; ++nt) {
                float pe = __expf(pv[nt][r] - mn);
                pv[nt][r] = pe;
                rs += pe;
            }
            rs = red_add16(rs);
            lrow[r] = lrow[r] * alpha + rs;
            #pragma unroll
            for (int j = 0; j < 4; ++j) o_[j][r] *= alpha;
        }

        // ---- C-layout P -> LDS -> A-fragment layout
        #pragma unroll
        for (int r = 0; r < 8; ++r)
            #pragma unroll
            for (int nt = 0; nt < 2; ++nt)
                Pl[(r + 8 * lh) * LSA + nt * 16 + ln] = (__bf16)pv[nt][r];
        const __bf16* pp = &Pl[ln * LSA + lh * 16];
        bf16x16 pf = pack16(*(const uint4*)pp, *(const uint4*)(pp + 8));

        // ---- wait for current V block (older TDM completes in order)
        if (more) __builtin_amdgcn_s_wait_tensorcnt(1);
        else      __builtin_amdgcn_s_wait_tensorcnt(0);
        asm volatile("" ::: "memory");   // keep LDS gathers below the wait

        // ---- P(16x32) @ V(32x64): gather V B-fragments from LDS, 4 WMMAs
        const __bf16* vsrc = Vs[p];
        #pragma unroll
        for (int nt = 0; nt < 4; ++nt) {
            union { __bf16 u[16]; bf16x16 v; } vv;
            #pragma unroll
            for (int j = 0; j < 16; ++j)
                vv.u[j] = vsrc[(lh * 16 + j) * VSA + nt * 16 + ln];
            o_[nt] = __builtin_amdgcn_wmma_f32_16x16x32_bf16(
                false, pf, false, vv.v, (short)0, o_[nt], false, false);
        }
        p ^= 1;
    }

    // ---- normalize and store bf16
    #pragma unroll
    for (int r = 0; r < 8; ++r) {
        const float inv = 1.0f / lrow[r];
        const int row = q0 + r + 8 * lh;
        #pragma unroll
        for (int nt = 0; nt < 4; ++nt)
            O[((size_t)b * SN + row) * DIMN + h * HDN + nt * 16 + ln] =
                (__bf16)(o_[nt][r] * inv);
    }
}

// ---------------------------------------------------------------------------
// Orchestration
// ---------------------------------------------------------------------------
extern "C" void kernel_launch(void* const* d_in, const int* in_sizes, int n_in,
                              void* d_out, int out_size, void* d_ws, size_t ws_size,
                              hipStream_t stream) {
    const float* x    = (const float*)d_in[0];
    // d_in[1] = mask (recomputed analytically: causal with -1e9 fill)
    const float* wq   = (const float*)d_in[2];
    const float* wk   = (const float*)d_in[3];
    const float* wv   = (const float*)d_in[4];
    const float* wo   = (const float*)d_in[5];
    const float* w1   = (const float*)d_in[6];
    const float* w2   = (const float*)d_in[7];
    const float* ln1w = (const float*)d_in[8];
    const float* ln1b = (const float*)d_in[9];
    const float* ln2w = (const float*)d_in[10];
    const float* ln2b = (const float*)d_in[11];
    float* out = (float*)d_out;

    char* ws = (char*)d_ws;
    const size_t MB = 1024 * 1024;
    __bf16* xn   = (__bf16*)(ws);            // 16MB (reused as hn)
    __bf16* qb   = (__bf16*)(ws + 16 * MB);  // 16MB
    __bf16* kbuf = (__bf16*)(ws + 32 * MB);  // 16MB
    __bf16* vbuf = (__bf16*)(ws + 48 * MB);  // 16MB
    __bf16* ab   = (__bf16*)(ws + 64 * MB);  // 16MB
    float*  hb   = (float*)(ws + 80 * MB);   // 32MB
    __bf16* ff1  = qb;  // reuse 64MB region [16MB,80MB) after attention

    dim3 blk256(256), blk32(32);
    dim3 gproj(DIMN / TN, ROWS / TM);
    dim3 gffn1(HIDN / TN, ROWS / TM);
    dim3 gattn(SN / 16, NHN, BN);

    // pre-norm attention
    ln_bf16<<<ROWS, blk256, 0, stream>>>(x, ln1w, ln1b, xn);
    gemm_bf16<0><<<gproj, blk256, 0, stream>>>(xn, wq, nullptr, qb,   ROWS, DIMN, DIMN);
    gemm_bf16<0><<<gproj, blk256, 0, stream>>>(xn, wk, nullptr, kbuf, ROWS, DIMN, DIMN);
    gemm_bf16<0><<<gproj, blk256, 0, stream>>>(xn, wv, nullptr, vbuf, ROWS, DIMN, DIMN);
    attn_flash<<<gattn, blk32, 0, stream>>>(qb, kbuf, vbuf, ab);
    gemm_bf16<2><<<gproj, blk256, 0, stream>>>(ab, wo, x, hb, ROWS, DIMN, DIMN);

    // pre-norm ReLU FFN
    ln_bf16<<<ROWS, blk256, 0, stream>>>(hb, ln2w, ln2b, xn);
    gemm_bf16<1><<<gffn1, blk256, 0, stream>>>(xn, w1, nullptr, ff1, ROWS, HIDN, DIMN);
    gemm_bf16<2><<<gproj, blk256, 0, stream>>>(ff1, w2, hb, out, ROWS, DIMN, HIDN);

    (void)in_sizes; (void)n_in; (void)out_size; (void)ws_size;
}
